// SteerableCNN_QM9_40742059770461
// MI455X (gfx1250) — matrix operand order repaired and checked by reading, not compiled, not run on hardware
//
#include <hip/hip_runtime.h>
#include <hip/hip_bf16.h>

// ---------------------------------------------------------------------------
// Steerable GNN (QM9-style) forward on MI455X (gfx1250, wave32, WMMA).
// - All WMMA B operands pre-packed into fragment-native layout (32 lanes x 16
//   halves contiguous): one 32B vector load per lane per fragment.
// - Edge kernel stages packed W2 fragments + MLP params in LDS.
// - Unsigned 32-bit hot-path indexing -> SADDR + voffset addressing.
// - Node state fused as sv[N][32][4] = {s, vx, vy, vz}: every per-edge gather
//   is a single b128; accumulators fused the same way (msum[N][32][4]).
// ---------------------------------------------------------------------------

typedef __attribute__((ext_vector_type(16))) _Float16 v16h;
typedef __attribute__((ext_vector_type(8)))  float    v8f;

#define Nn    50000
#define Ee    800000
#define Ff    32
#define Hh    64
#define NLAY  4
#define NINVc 128
#define Gg    2048

__device__ inline float eluf(float x) { return x > 0.f ? x : (expf(x) - 1.f); }

__device__ inline v8f wmma16(v16h a, v16h b, v8f c) {
  // D = A(16x32 f16) * B(32x16 f16) + C(16x16 f32)
  return __builtin_amdgcn_wmma_f32_16x16x32_f16(
      /*neg_a=*/false, a, /*neg_b=*/false, b,
      /*c_mod=*/(short)0, c, /*reuse_a=*/false, /*reuse_b=*/false);
}

// Load one pre-packed B fragment (512 halves): lane-major, 16 halves per lane.
__device__ inline v16h load_frag(const _Float16* __restrict__ frags, unsigned frag,
                                 unsigned lane) {
  return *(const v16h*)(frags + (frag << 9) + (lane << 4));
}
__device__ inline v16h lds_frag(const _Float16* sf, unsigned frag, unsigned lane) {
  return *(const v16h*)(sf + (frag << 9) + (lane << 4));
}

// ---------------------------------------------------------------------------
// Pack a row-major f32 weight W[K x Ncols] into WMMA-B fragment layout (f16).
// ---------------------------------------------------------------------------
__global__ void pack_b_kernel(const float* __restrict__ W, _Float16* __restrict__ out,
                              int K, int Ncols) {
  int total = K * Ncols;
  int i = blockIdx.x * blockDim.x + threadIdx.x;
  if (i >= total) return;
  int ntiles = Ncols >> 4;
  int frag   = i >> 9;
  int within = i & 511;
  int lane   = within >> 4;
  int slot   = within & 15;
  int kt = frag / ntiles;
  int nt = frag - kt * ntiles;
  int n  = lane & 15;
  int kb = (lane >> 4) << 3;
  int k  = kt * 32 + (slot < 8 ? kb + slot : 16 + kb + (slot - 8));
  out[i] = (_Float16)W[(size_t)k * Ncols + nt * 16 + n];
}

// ---------------------------------------------------------------------------
// Small prep kernels
// ---------------------------------------------------------------------------

__global__ void edge_geom_kernel(const float* __restrict__ pos,
                                 const int* __restrict__ src,
                                 const int* __restrict__ dst,
                                 float* __restrict__ d_e, float* __restrict__ u_e,
                                 float* __restrict__ cnt, int E) {
  unsigned e = blockIdx.x * blockDim.x + threadIdx.x;
  if (e >= (unsigned)E) return;
  unsigned s0 = (unsigned)src[e], d0 = (unsigned)dst[e];
  float dx = pos[d0 * 3u + 0] - pos[s0 * 3u + 0];
  float dy = pos[d0 * 3u + 1] - pos[s0 * 3u + 1];
  float dz = pos[d0 * 3u + 2] - pos[s0 * 3u + 2];
  float dd = sqrtf(dx * dx + dy * dy + dz * dz + 1e-6f);
  d_e[e] = dd;
  float inv = 1.0f / dd;
  u_e[e * 3u + 0] = dx * inv;
  u_e[e * 3u + 1] = dy * inv;
  u_e[e * 3u + 2] = dz * inv;
  atomicAdd(&cnt[d0], 1.0f);
}

__global__ void clamp1_kernel(float* __restrict__ c, int n) {
  unsigned i = blockIdx.x * blockDim.x + threadIdx.x;
  if (i < (unsigned)n) c[i] = fmaxf(c[i], 1.0f);
}

// Embedding into fused layout: sv[(n*32+f)*4] = {s, vx, vy, vz}
__global__ void embed_kernel(const float* __restrict__ x, const float* __restrict__ pos,
                             const float* __restrict__ W_es, const float* __restrict__ b_es,
                             const float* __restrict__ W_ev,
                             float* __restrict__ sv, int N) {
  unsigned i = blockIdx.x * blockDim.x + threadIdx.x;
  if (i >= (unsigned)(N * Ff)) return;
  unsigned n = i >> 5, f = i & 31;
  float a = b_es[f], bv = 0.f;
#pragma unroll
  for (unsigned k = 0; k < 5; ++k) {
    float xv = x[n * 5u + k];
    a  += xv * W_es[k * Ff + f];
    bv += xv * W_ev[k * Ff + f];
  }
  float4 o;
  o.x = a;
  o.y = bv * pos[n * 3u + 0];
  o.z = bv * pos[n * 3u + 1];
  o.w = bv * pos[n * 3u + 2];
  *(float4*)(sv + i * 4u) = o;
}

__global__ void gcnt_kernel(const int* __restrict__ batch, float* __restrict__ gcnt, int N) {
  unsigned i = blockIdx.x * blockDim.x + threadIdx.x;
  if (i < (unsigned)N) atomicAdd(&gcnt[(unsigned)batch[i]], 1.0f);
}

// ---------------------------------------------------------------------------
// Per-layer edge kernel: edge MLP (WMMA) + message formation + atomic scatter.
// One wave handles a tile of 16 edges. W2 fragments + W1/b1/b2 staged in LDS.
// ---------------------------------------------------------------------------
__global__ void __launch_bounds__(256) edge_layer_kernel(
    const float* __restrict__ d_e, const float* __restrict__ u_e,
    const float* __restrict__ eattr,
    const int* __restrict__ src, const int* __restrict__ dst,
    const float* __restrict__ sv,                                   // fused node state
    const float* __restrict__ W1, const float* __restrict__ b1,     // [5,64],[64]
    const _Float16* __restrict__ W2p, const float* __restrict__ b2, // packed 12 frags, [96]
    float* __restrict__ msum, int E) {                              // fused accumulators
  __shared__ _Float16 sW2[12 * 512];   // 12 KB
  __shared__ float    sPar[480];       // W1(320) | b1(64) | b2(96)

  {
    const uint4* gs4 = (const uint4*)W2p;
    uint4* ds4 = (uint4*)sW2;
    for (unsigned i = threadIdx.x; i < (12 * 512) / 8; i += blockDim.x) ds4[i] = gs4[i];
    for (unsigned i = threadIdx.x; i < 480; i += blockDim.x) {
      float val;
      if (i < 320)      val = W1[i];
      else if (i < 384) val = b1[i - 320];
      else              val = b2[i - 384];
      sPar[i] = val;
    }
  }
  __syncthreads();
  const float* sW1 = sPar;
  const float* sb1 = sPar + 320;
  const float* sb2 = sPar + 384;

  unsigned lane  = threadIdx.x & 31;
  unsigned tile  = blockIdx.x * (blockDim.x >> 5) + (threadIdx.x >> 5);
  unsigned ebase = tile << 4;
  if (ebase >= (unsigned)E) return;

  unsigned e0 = ebase + (lane & 15);
  float ein[5];
  ein[0] = d_e[e0];
  {
    float4 ea = *(const float4*)(eattr + e0 * 4u);
    ein[1] = ea.x; ein[2] = ea.y; ein[3] = ea.z; ein[4] = ea.w;
  }

  // h1 = elu(ein @ W1 + b1), built directly in A-fragment slot order:
  // slot i<8  -> hidden col kb+i ; slot i>=8 -> col 16+kb+(i-8)  (kb = 0|8)
  unsigned kb = (lane >> 4) << 3;
  float hA[16], hB[16];
#pragma unroll
  for (int i = 0; i < 8; ++i) {
    hA[i]     = sb1[kb + i];
    hA[8 + i] = sb1[16 + kb + i];
    hB[i]     = sb1[32 + kb + i];
    hB[8 + i] = sb1[48 + kb + i];
  }
#pragma unroll
  for (int w = 0; w < 5; ++w) {
    float xv = ein[w];
    const float* pw = sW1 + w * 64;
#pragma unroll
    for (int i = 0; i < 8; ++i) {
      hA[i]     += xv * pw[kb + i];
      hA[8 + i] += xv * pw[16 + kb + i];
      hB[i]     += xv * pw[32 + kb + i];
      hB[8 + i] += xv * pw[48 + kb + i];
    }
  }
  v16h a0 = {}, a1 = {};
#pragma unroll
  for (int i = 0; i < 16; ++i) {
    a0[i] = (_Float16)eluf(hA[i]);
    a1[i] = (_Float16)eluf(hB[i]);
  }

  unsigned moff = (lane >> 4) << 3;  // C-fragment row offset (0 or 8)
  unsigned n    = lane & 15;

  // All 6 accumulators: acc[ntp*3 + {gs,gv,gsv}]
  v8f acc[6];
#pragma unroll
  for (int ntp = 0; ntp < 2; ++ntp) {
    v8f cgs = {}, cgv = {}, cgsv = {};
    v16h b;
    b = lds_frag(sW2, 0 * 6 + ntp,     lane);  cgs  = wmma16(a0, b, cgs);
    b = lds_frag(sW2, 1 * 6 + ntp,     lane);  cgs  = wmma16(a1, b, cgs);
    b = lds_frag(sW2, 0 * 6 + 2 + ntp, lane);  cgv  = wmma16(a0, b, cgv);
    b = lds_frag(sW2, 1 * 6 + 2 + ntp, lane);  cgv  = wmma16(a1, b, cgv);
    b = lds_frag(sW2, 0 * 6 + 4 + ntp, lane);  cgsv = wmma16(a0, b, cgsv);
    b = lds_frag(sW2, 1 * 6 + 4 + ntp, lane);  cgsv = wmma16(a1, b, cgsv);
    acc[ntp * 3 + 0] = cgs;
    acc[ntp * 3 + 1] = cgv;
    acc[ntp * 3 + 2] = cgsv;
  }

  // This lane's 8 C-rows are edges ebase+moff .. +7 (contiguous): hoist the
  // per-edge metadata as vector loads.
  int srcs[8], dsts[8];
  {
    const int4* ps = (const int4*)(src + ebase + moff);
    *(int4*)(srcs + 0) = ps[0];
    *(int4*)(srcs + 4) = ps[1];
    const int4* pd = (const int4*)(dst + ebase + moff);
    *(int4*)(dsts + 0) = pd[0];
    *(int4*)(dsts + 4) = pd[1];
  }
  float uu[24];
  {
    const float4* pu = (const float4*)(u_e + (ebase + moff) * 3u);
#pragma unroll
    for (int q = 0; q < 6; ++q) *(float4*)(uu + 4 * q) = pu[q];
  }

#pragma unroll
  for (int r = 0; r < 8; ++r) {
    unsigned sr = (unsigned)srcs[r];
    unsigned dr = (unsigned)dsts[r];
    float ux = uu[3 * r + 0], uy = uu[3 * r + 1], uz = uu[3 * r + 2];
#pragma unroll
    for (int ntp = 0; ntp < 2; ++ntp) {
      unsigned fc = (unsigned)(ntp * 16) + n;
      float gs  = acc[ntp * 3 + 0][r] + sb2[fc];
      float gv  = acc[ntp * 3 + 1][r] + sb2[32 + fc];
      float gsv = acc[ntp * 3 + 2][r] + sb2[64 + fc];
      // Fused gather: {s, vx, vy, vz} in one b128.
      float4 sv4 = *(const float4*)(sv + (sr * 32u + fc) * 4u);
      float ms = gs * sv4.x;
      float kk = gsv * sv4.x;
      float mvx = gv * sv4.y + kk * ux;
      float mvy = gv * sv4.z + kk * uy;
      float mvz = gv * sv4.w + kk * uz;
      float* ap = msum + (dr * 32u + fc) * 4u;
      atomicAdd(ap + 0, ms);
      atomicAdd(ap + 1, mvx);
      atomicAdd(ap + 2, mvy);
      atomicAdd(ap + 3, mvz);
    }
  }
}

// ---------------------------------------------------------------------------
// Per-layer node update: s += elu((ms/deg) @ Ws); v += (mv/deg) @ Wv.
// One wave handles 16 nodes; K=32 -> one WMMA per 16x16 tile. Fused-layout
// rows load as 8 contiguous float4s whose components are the fragment slots.
// ---------------------------------------------------------------------------
__global__ void __launch_bounds__(256) node_update_kernel(
    const float* __restrict__ msum, const float* __restrict__ denom,
    const _Float16* __restrict__ Wsp, const _Float16* __restrict__ Wvp, // packed, 2 frags each
    float* __restrict__ sv, int N) {
  unsigned lane  = threadIdx.x & 31;
  unsigned tile  = blockIdx.x * (blockDim.x >> 5) + (threadIdx.x >> 5);
  unsigned nbase = tile << 4;
  if (nbase >= (unsigned)N) return;

  unsigned node = nbase + (lane & 15);
  float inv_dn = 1.0f / denom[node];
  unsigned kb = (lane >> 4) << 3;

  float4 q[16];
  {
    const float4* p0 = (const float4*)(msum + (node * 32u + kb) * 4u);
#pragma unroll
    for (int i = 0; i < 8; ++i) q[i] = p0[i];
    const float4* p1 = (const float4*)(msum + (node * 32u + 16u + kb) * 4u);
#pragma unroll
    for (int i = 0; i < 8; ++i) q[8 + i] = p1[i];
  }
  v16h aS = {}, aX = {}, aY = {}, aZ = {};
#pragma unroll
  for (int i = 0; i < 16; ++i) {
    aS[i] = (_Float16)(q[i].x * inv_dn);
    aX[i] = (_Float16)(q[i].y * inv_dn);
    aY[i] = (_Float16)(q[i].z * inv_dn);
    aZ[i] = (_Float16)(q[i].w * inv_dn);
  }

  unsigned moff = (lane >> 4) << 3;
  unsigned n    = lane & 15;
#pragma unroll
  for (int nt = 0; nt < 2; ++nt) {
    v16h bs = load_frag(Wsp, nt, lane);
    v16h bv = load_frag(Wvp, nt, lane);
    v8f cs = {}, cx = {}, cy = {}, cz = {};
    cs = wmma16(aS, bs, cs);
    cx = wmma16(aX, bv, cx);
    cy = wmma16(aY, bv, cy);
    cz = wmma16(aZ, bv, cz);
#pragma unroll
    for (int r = 0; r < 8; ++r) {
      unsigned nd = nbase + r + moff;
      unsigned f  = (unsigned)(nt * 16) + n;
      float4* p = (float4*)(sv + (nd * 32u + f) * 4u);
      float4 cur = *p;
      cur.x += eluf(cs[r]);
      cur.y += cx[r];
      cur.z += cy[r];
      cur.w += cz[r];
      *p = cur;
    }
  }
}

// ---------------------------------------------------------------------------
// Invariant map + global pooling: inv = [s, |v|] @ W_inv + b_inv scattered
// into per-graph sums. One wave handles 16 nodes (K=64 -> 2 WMMA k-steps).
// ---------------------------------------------------------------------------
__global__ void __launch_bounds__(256) inv_pool_kernel(
    const float* __restrict__ sv,
    const int* __restrict__ batch,
    const _Float16* __restrict__ Winvp,  // packed 16 frags
    const float* __restrict__ b_inv,
    float* __restrict__ xgsum, int N) {
  unsigned lane  = threadIdx.x & 31;
  unsigned tile  = blockIdx.x * (blockDim.x >> 5) + (threadIdx.x >> 5);
  unsigned nbase = tile << 4;
  if (nbase >= (unsigned)N) return;

  unsigned node = nbase + (lane & 15);
  unsigned kb = (lane >> 4) << 3;

  float4 q[16];
  {
    const float4* p0 = (const float4*)(sv + (node * 32u + kb) * 4u);
#pragma unroll
    for (int i = 0; i < 8; ++i) q[i] = p0[i];
    const float4* p1 = (const float4*)(sv + (node * 32u + 16u + kb) * 4u);
#pragma unroll
    for (int i = 0; i < 8; ++i) q[8 + i] = p1[i];
  }
  v16h a0 = {}, a1 = {};
#pragma unroll
  for (int i = 0; i < 16; ++i) {
    a0[i] = (_Float16)q[i].x;
    a1[i] = (_Float16)sqrtf(q[i].y * q[i].y + q[i].z * q[i].z + q[i].w * q[i].w + 1e-6f);
  }

  // Per-lane graph ids for this lane's 8 C-rows (contiguous nodes).
  int gids[8];
  {
    const int4* pb = (const int4*)(batch + nbase + ((lane >> 4) << 3));
    *(int4*)(gids + 0) = pb[0];
    *(int4*)(gids + 4) = pb[1];
  }

  unsigned n = lane & 15;
#pragma unroll
  for (int nt = 0; nt < 8; ++nt) {
    v8f c = {};
    v16h b = load_frag(Winvp, 0 * 8 + nt, lane);
    c = wmma16(a0, b, c);
    b = load_frag(Winvp, 1 * 8 + nt, lane);
    c = wmma16(a1, b, c);
#pragma unroll
    for (int r = 0; r < 8; ++r) {
      unsigned col = (unsigned)(nt * 16) + n;
      float val = c[r] + b_inv[col];
      atomicAdd(&xgsum[(unsigned)gids[r] * 128u + col], val);
    }
  }
}

__global__ void pool_div_kernel(const float* __restrict__ xgsum,
                                const float* __restrict__ gcnt,
                                float* __restrict__ xg, int n) {
  unsigned i = blockIdx.x * blockDim.x + threadIdx.x;
  if (i < (unsigned)n) xg[i] = xgsum[i] / fmaxf(gcnt[i >> 7], 1.0f);
}

// BN stats per column: scale = g*rsqrt(var+1e-5), shift = be - mean*scale.
__global__ void bnstats_kernel(const float* __restrict__ X,
                               const float* __restrict__ g, const float* __restrict__ be,
                               float* __restrict__ scale, float* __restrict__ shift,
                               int rows) {
  __shared__ float ssum[256], ssq[256];
  unsigned c = blockIdx.x;
  float a = 0.f, b = 0.f;
  for (unsigned r = threadIdx.x; r < (unsigned)rows; r += blockDim.x) {
    float x = X[r * (unsigned)NINVc + c];
    a += x;
    b += x * x;
  }
  ssum[threadIdx.x] = a;
  ssq[threadIdx.x]  = b;
  __syncthreads();
  for (int o = blockDim.x >> 1; o > 0; o >>= 1) {
    if ((int)threadIdx.x < o) {
      ssum[threadIdx.x] += ssum[threadIdx.x + o];
      ssq[threadIdx.x]  += ssq[threadIdx.x + o];
    }
    __syncthreads();
  }
  if (threadIdx.x == 0) {
    float mean = ssum[0] / rows;
    float var  = ssq[0] / rows - mean * mean;
    float rs   = rsqrtf(var + 1e-5f);
    float sc   = g[c] * rs;
    scale[c] = sc;
    shift[c] = be[c] - mean * sc;
  }
}

// z1 = elu(bn(xg)) @ Wf1 + bf1 via WMMA (K=128 -> 4 k-steps, 8 n-tiles).
__global__ void __launch_bounds__(256) head1_kernel(
    const float* __restrict__ xg,
    const float* __restrict__ scale, const float* __restrict__ shift,
    const _Float16* __restrict__ Wf1p, const float* __restrict__ bf1,
    float* __restrict__ z1, int G) {
  unsigned lane  = threadIdx.x & 31;
  unsigned tile  = blockIdx.x * (blockDim.x >> 5) + (threadIdx.x >> 5);
  unsigned rbase = tile << 4;
  if (rbase >= (unsigned)G) return;

  unsigned row = rbase + (lane & 15);
  unsigned kb = (lane >> 4) << 3;
  v16h A[4];
#pragma unroll
  for (int kf = 0; kf < 4; ++kf) {
    v16h a = {};
#pragma unroll
    for (int i = 0; i < 8; ++i) {
      unsigned k0 = (unsigned)(kf * 32) + kb + i;
      unsigned k1 = (unsigned)(kf * 32) + 16u + kb + i;
      float t0 = xg[row * 128u + k0] * scale[k0] + shift[k0];
      float t1 = xg[row * 128u + k1] * scale[k1] + shift[k1];
      a[i]     = (_Float16)eluf(t0);
      a[8 + i] = (_Float16)eluf(t1);
    }
    A[kf] = a;
  }

  unsigned moff = (lane >> 4) << 3;
  unsigned n    = lane & 15;
#pragma unroll
  for (int nt = 0; nt < 8; ++nt) {
    v8f c = {};
#pragma unroll
    for (int kf = 0; kf < 4; ++kf) {
      v16h b = load_frag(Wf1p, kf * 8 + nt, lane);
      c = wmma16(A[kf], b, c);
    }
#pragma unroll
    for (int r = 0; r < 8; ++r) {
      unsigned rr  = rbase + r + moff;
      unsigned col = (unsigned)(nt * 16) + n;
      z1[rr * 128u + col] = c[r] + bf1[col];
    }
  }
}

// out[g] = sum_c elu(bn(z1))[g,c] * Wf2[c] + bf2
__global__ void head2_kernel(const float* __restrict__ z1,
                             const float* __restrict__ scale, const float* __restrict__ shift,
                             const float* __restrict__ Wf2, const float* __restrict__ bf2,
                             float* __restrict__ out) {
  __shared__ float red[128];
  unsigned g = blockIdx.x;
  unsigned c = threadIdx.x;
  float t = z1[g * 128u + c] * scale[c] + shift[c];
  red[c] = eluf(t) * Wf2[c];
  __syncthreads();
  for (int o = 64; o > 0; o >>= 1) {
    if ((int)c < o) red[c] += red[c + o];
    __syncthreads();
  }
  if (c == 0) out[g] = red[0] + bf2[0];
}

// ---------------------------------------------------------------------------
// Host launcher
// ---------------------------------------------------------------------------
extern "C" void kernel_launch(void* const* d_in, const int* in_sizes, int n_in,
                              void* d_out, int out_size, void* d_ws, size_t ws_size,
                              hipStream_t stream) {
  (void)in_sizes; (void)n_in; (void)out_size; (void)ws_size;

  const float* x     = (const float*)d_in[0];
  const float* pos   = (const float*)d_in[1];
  const int*   eidx  = (const int*)d_in[2];
  const float* eattr = (const float*)d_in[3];
  const int*   batch = (const int*)d_in[4];
  const float* W_es  = (const float*)d_in[5];
  const float* b_es  = (const float*)d_in[6];
  const float* W_ev  = (const float*)d_in[7];
  const float* W1    = (const float*)d_in[8];
  const float* b1    = (const float*)d_in[9];
  const float* W2    = (const float*)d_in[10];
  const float* b2    = (const float*)d_in[11];
  const float* Ws    = (const float*)d_in[12];
  const float* Wv    = (const float*)d_in[13];
  const float* W_inv = (const float*)d_in[14];
  const float* b_inv = (const float*)d_in[15];
  const float* g1    = (const float*)d_in[16];
  const float* be1   = (const float*)d_in[17];
  const float* Wf1   = (const float*)d_in[18];
  const float* bf1   = (const float*)d_in[19];
  const float* g2    = (const float*)d_in[20];
  const float* be2   = (const float*)d_in[21];
  const float* Wf2   = (const float*)d_in[22];
  const float* bf2   = (const float*)d_in[23];

  const int* src = eidx;
  const int* dst = eidx + Ee;

  // Workspace carve-out (256B aligned slices).
  char* base = (char*)d_ws;
  size_t off = 0;
  auto carve = [&](size_t bytes) -> void* {
    void* p = base + off;
    off += (bytes + 255) & ~(size_t)255;
    return p;
  };
  float* d_e    = (float*)carve((size_t)Ee * 4);
  float* u_e    = (float*)carve((size_t)Ee * 12);
  float* svb    = (float*)carve((size_t)Nn * Ff * 16);   // fused {s,vx,vy,vz}
  float* msum   = (float*)carve((size_t)Nn * Ff * 16);   // fused accumulators
  float* denom  = (float*)carve((size_t)Nn * 4);
  float* xgsum  = (float*)carve((size_t)Gg * NINVc * 4);
  float* gcnt   = (float*)carve((size_t)Gg * 4);
  float* xg     = (float*)carve((size_t)Gg * NINVc * 4);
  float* z1b    = (float*)carve((size_t)Gg * NINVc * 4);
  float* scale1 = (float*)carve(NINVc * 4);
  float* shift1 = (float*)carve(NINVc * 4);
  float* scale2 = (float*)carve(NINVc * 4);
  float* shift2 = (float*)carve(NINVc * 4);
  _Float16* W2p   = (_Float16*)carve((size_t)NLAY * Hh * 96 * 2);
  _Float16* Wsp   = (_Float16*)carve((size_t)NLAY * Ff * Ff * 2);
  _Float16* Wvp   = (_Float16*)carve((size_t)NLAY * Ff * Ff * 2);
  _Float16* Winvp = (_Float16*)carve((size_t)2 * Ff * NINVc * 2);
  _Float16* Wf1p  = (_Float16*)carve((size_t)NINVc * NINVc * 2);

  hipMemsetAsync(denom, 0, (size_t)Nn * 4, stream);
  hipMemsetAsync(xgsum, 0, (size_t)Gg * NINVc * 4, stream);
  hipMemsetAsync(gcnt, 0, (size_t)Gg * 4, stream);

  // Pack all WMMA B operands into fragment-native f16 layout.
  for (int l = 0; l < NLAY; ++l) {
    pack_b_kernel<<<(Hh * 96 + 255) / 256, 256, 0, stream>>>(
        W2 + (size_t)l * Hh * 96, W2p + (size_t)l * Hh * 96, Hh, 96);
    pack_b_kernel<<<(Ff * Ff + 255) / 256, 256, 0, stream>>>(
        Ws + (size_t)l * Ff * Ff, Wsp + (size_t)l * Ff * Ff, Ff, Ff);
    pack_b_kernel<<<(Ff * Ff + 255) / 256, 256, 0, stream>>>(
        Wv + (size_t)l * Ff * Ff, Wvp + (size_t)l * Ff * Ff, Ff, Ff);
  }
  pack_b_kernel<<<(2 * Ff * NINVc + 255) / 256, 256, 0, stream>>>(W_inv, Winvp, 2 * Ff, NINVc);
  pack_b_kernel<<<(NINVc * NINVc + 255) / 256, 256, 0, stream>>>(Wf1, Wf1p, NINVc, NINVc);

  edge_geom_kernel<<<(Ee + 255) / 256, 256, 0, stream>>>(pos, src, dst, d_e, u_e, denom, Ee);
  clamp1_kernel<<<(Nn + 255) / 256, 256, 0, stream>>>(denom, Nn);
  embed_kernel<<<(Nn * Ff + 255) / 256, 256, 0, stream>>>(x, pos, W_es, b_es, W_ev, svb, Nn);

  int edge_tiles = Ee / 16;             // 50000
  int node_tiles = Nn / 16;             // 3125
  for (int l = 0; l < NLAY; ++l) {
    hipMemsetAsync(msum, 0, (size_t)Nn * Ff * 16, stream);
    edge_layer_kernel<<<(edge_tiles + 7) / 8, 256, 0, stream>>>(
        d_e, u_e, eattr, src, dst, svb,
        W1 + (size_t)l * 5 * Hh, b1 + (size_t)l * Hh,
        W2p + (size_t)l * Hh * 96, b2 + (size_t)l * 96,
        msum, Ee);
    node_update_kernel<<<(node_tiles + 7) / 8, 256, 0, stream>>>(
        msum, denom, Wsp + (size_t)l * Ff * Ff, Wvp + (size_t)l * Ff * Ff,
        svb, Nn);
  }

  gcnt_kernel<<<(Nn + 255) / 256, 256, 0, stream>>>(batch, gcnt, Nn);
  inv_pool_kernel<<<(node_tiles + 7) / 8, 256, 0, stream>>>(
      svb, batch, Winvp, b_inv, xgsum, Nn);
  pool_div_kernel<<<(Gg * NINVc + 255) / 256, 256, 0, stream>>>(xgsum, gcnt, xg, Gg * NINVc);

  bnstats_kernel<<<NINVc, 256, 0, stream>>>(xg, g1, be1, scale1, shift1, Gg);
  head1_kernel<<<(Gg / 16 + 7) / 8, 256, 0, stream>>>(xg, scale1, shift1, Wf1p, bf1, z1b, Gg);
  bnstats_kernel<<<NINVc, 256, 0, stream>>>(z1b, g2, be2, scale2, shift2, Gg);
  head2_kernel<<<Gg, 128, 0, stream>>>(z1b, scale2, shift2, Wf2, bf2, (float*)d_out);
}
